// RGCN_8280696947368
// MI455X (gfx1250) — matrix operand order repaired and checked by reading, not compiled, not run on hardware
//
#include <hip/hip_runtime.h>
#include <hip/hip_bf16.h>
#include <stdint.h>

#define NNODES 100000
#define NEDGES 1600000
#define NREL   4
#define NGRAPH 512
#define NCLS   10
#define NBLK   (NNODES / 16)   // 6250 row blocks of 16

typedef __attribute__((ext_vector_type(16))) __bf16 v16bf;
typedef __attribute__((ext_vector_type(8)))  float  v8f;

__device__ __forceinline__ unsigned short f2bf(float f) {
  union { float f; unsigned int u; } v; v.f = f;
  unsigned int u = v.u;
  // round-to-nearest-even fp32 -> bf16
  return (unsigned short)((u + 0x7FFFu + ((u >> 16) & 1u)) >> 16);
}

__global__ void k_zero(unsigned int* p, long n) {
  long i = (long)blockIdx.x * blockDim.x + threadIdx.x;
  if (i < n) p[i] = 0u;
}

// per-(dst, relation) edge counts (layer-invariant; mean normalization)
__global__ void k_count(const int* __restrict__ dstv, const int* __restrict__ et,
                        int* __restrict__ cnt, int n) {
  int i = blockIdx.x * blockDim.x + threadIdx.x;
  if (i < n) atomicAdd(&cnt[(long)dstv[i] * NREL + et[i]], 1);
}

__global__ void k_bcount(const int* __restrict__ batch, int* __restrict__ bc, int n) {
  int i = blockIdx.x * blockDim.x + threadIdx.x;
  if (i < n) atomicAdd(&bc[batch[i]], 1);
}

// Pack all (3 layers x (4 relations + root)) 64x64 weights into the WMMA
// B-fragment layout (16-bit B 32x16, contiguous K order used here): lanes 0-15 hold K=0..15 (elem i -> K=i),
// lanes 16-31 hold K=16..31; N = lane%16; two K-blocks (kb) cover K=64.
__global__ void k_pack_w(const float* __restrict__ W1, const float* __restrict__ root1,
                         const float* __restrict__ Wl, const float* __restrict__ rootl,
                         unsigned short* __restrict__ wfrag) {
  int tid = blockIdx.x * blockDim.x + threadIdx.x;
  if (tid >= 3 * 5 * 4096) return;
  int layer = tid / (5 * 4096);
  int rem   = tid % (5 * 4096);
  int w = rem / 4096;
  int e = rem % 4096;
  int k = e >> 6, n = e & 63;               // row-major [K][N]
  const float* src;
  if (layer == 0) src = (w < 4) ? (W1 + (long)w * 4096) : root1;
  else            src = (w < 4) ? (Wl + (long)((layer - 1) * 4 + w) * 4096)
                                : (rootl + (long)(layer - 1) * 4096);
  float val = src[e];
  int t = n >> 4, nl = n & 15;
  int kb = k >> 5, kl = k & 31;
  int lhi = kl >> 4, ii = kl & 15;
  int lane = lhi * 16 + nl;
  long idx = (((((long)layer * 5 + w) * 4 + t) * 2 + kb) * 32 + lane) * 16 + ii;
  wfrag[idx] = f2bf(val);
}

// Pack activations into WMMA A-fragment layout (16-bit A 16x32):
// lane = (k-half)*16 + row%16 ; elem i -> K = 16*(k-half) + i (matches B pack).
// Optional fused ReLU and optional fp32 writeback (for final layer / pooling).
__global__ void k_pack_a(const float* __restrict__ src, unsigned short* __restrict__ afrag,
                         float* wb, int do_relu, int do_pack, int do_wb, long n) {
  long tid = (long)blockIdx.x * blockDim.x + threadIdx.x;
  if (tid >= n) return;
  float v = src[tid];
  if (do_relu) v = fmaxf(v, 0.0f);
  if (do_wb) wb[tid] = v;
  if (do_pack) {
    int nd = (int)(tid >> 6);
    int k  = (int)(tid & 63);
    int m0 = nd >> 4, lrow = nd & 15;
    int kb = k >> 5, kk = k & 31;
    int lhi = kk >> 4;
    int ii  = kk & 15;  // contiguous K map: K = 16*lhi + ii
    int lane = lhi * 16 + lrow;
    long idx = (((long)m0 * 2 + kb) * 32 + lane) * 16 + ii;
    afrag[idx] = f2bf(v);
  }
}

// One wave computes a 16x64 output slab for one of the 5 weight matrices.
// w<4 -> h_all[w]; w==4 -> root path into `out` with bias (accumulator init).
__global__ void __launch_bounds__(256)
k_gemm(const unsigned short* __restrict__ afrag,
       const unsigned short* __restrict__ wfrag,   // this layer's fragment base
       const float* __restrict__ bias,
       float* __restrict__ h_all, float* __restrict__ out) {
  int wave = (blockIdx.x * blockDim.x + threadIdx.x) >> 5;
  int lane = threadIdx.x & 31;
  int w  = wave % 5;
  int m0 = wave / 5;
  if (m0 >= NBLK) return;                    // wave-uniform; EXEC stays all-1s
  const unsigned short* ab = afrag + (long)m0 * 2 * 32 * 16;
  v16bf a0 = *(const v16bf*)(ab + (long)(0 * 32 + lane) * 16);
  v16bf a1 = *(const v16bf*)(ab + (long)(1 * 32 + lane) * 16);
  const unsigned short* wb = wfrag + (long)w * 4 * 2 * 32 * 16;
  float* dbase = (w < 4) ? (h_all + ((long)w * NNODES + (long)m0 * 16) * 64)
                         : (out + (long)m0 * 16 * 64);
  int ncol = lane & 15;
  int mh   = (lane >> 4) * 8;                // D layout: M = v + 8*(lane>=16)
#pragma unroll
  for (int t = 0; t < 4; ++t) {
    v16bf b0 = *(const v16bf*)(wb + ((long)(t * 2 + 0) * 32 + lane) * 16);
    v16bf b1 = *(const v16bf*)(wb + ((long)(t * 2 + 1) * 32 + lane) * 16);
    float cv = (w == 4) ? bias[t * 16 + ncol] : 0.0f;  // wave-uniform branch
    v8f c = {cv, cv, cv, cv, cv, cv, cv, cv};
    c = __builtin_amdgcn_wmma_f32_16x16x32_bf16(false, a0, false, b0,
                                                (short)0, c, false, false);
    c = __builtin_amdgcn_wmma_f32_16x16x32_bf16(false, a1, false, b1,
                                                (short)0, c, false, false);
#pragma unroll
    for (int v = 0; v < 8; ++v)
      dbase[(long)(mh + v) * 64 + (t * 16 + ncol)] = c[v];
  }
}

// 16 lanes per edge, float4 each: gather h_all[rel][src], mean-scale, atomic
// scatter-add into out[dst]. Whole working set is L2-resident on MI455X.
__global__ void k_edges(const int* __restrict__ srcv, const int* __restrict__ dstv,
                        const int* __restrict__ et, const int* __restrict__ cnt,
                        const float* __restrict__ h_all, float* __restrict__ out) {
  long tid = (long)blockIdx.x * blockDim.x + threadIdx.x;
  long e = tid >> 4;
  int  q = (int)(tid & 15);
  if (e >= NEDGES) return;
  int s = srcv[e], d = dstv[e], r = et[e];
  float c = (float)cnt[(long)d * NREL + r];
  float norm = 1.0f / fmaxf(c, 1.0f);
  float4 v = *(const float4*)(h_all + ((long)r * NNODES + s) * 64 + q * 4);
  float* o = out + (long)d * 64 + q * 4;
  atomicAdd(o + 0, v.x * norm);
  atomicAdd(o + 1, v.y * norm);
  atomicAdd(o + 2, v.z * norm);
  atomicAdd(o + 3, v.w * norm);
}

__global__ void k_pool(const float* __restrict__ h, const int* __restrict__ batch,
                       float* __restrict__ pools) {
  long tid = (long)blockIdx.x * blockDim.x + threadIdx.x;
  long nd = tid >> 4;
  int  q  = (int)(tid & 15);
  if (nd >= NNODES) return;
  int g = batch[nd];
  float4 v = *(const float4*)(h + nd * 64 + q * 4);
  float* o = pools + (long)g * 64 + q * 4;
  atomicAdd(o + 0, v.x); atomicAdd(o + 1, v.y);
  atomicAdd(o + 2, v.z); atomicAdd(o + 3, v.w);
}

__global__ void k_head1(const float* __restrict__ pools, const int* __restrict__ bc,
                        const float* __restrict__ w, const float* __restrict__ b,
                        float* __restrict__ g1) {
  int tid = blockIdx.x * blockDim.x + threadIdx.x;
  if (tid >= NGRAPH * 64) return;
  int g = tid >> 6, j = tid & 63;
  float inv = 1.0f / fmaxf((float)bc[g], 1.0f);
  float acc = b[j];
  for (int k = 0; k < 64; ++k) acc += pools[g * 64 + k] * inv * w[k * 64 + j];
  g1[tid] = fmaxf(acc, 0.0f);
}

__global__ void k_head2(const float* __restrict__ g1, const float* __restrict__ w,
                        const float* __restrict__ b, float* __restrict__ outp) {
  int tid = blockIdx.x * blockDim.x + threadIdx.x;
  if (tid >= NGRAPH * NCLS) return;
  int g = tid / NCLS, cc = tid % NCLS;
  float acc = b[cc];
  for (int k = 0; k < 64; ++k) acc += g1[g * 64 + k] * w[k * NCLS + cc];
  outp[tid] = acc;
}

extern "C" void kernel_launch(void* const* d_in, const int* in_sizes, int n_in,
                              void* d_out, int out_size, void* d_ws, size_t ws_size,
                              hipStream_t stream) {
  (void)in_sizes; (void)n_in; (void)out_size; (void)ws_size;
  const float* x     = (const float*)d_in[0];
  const int*   eidx  = (const int*)d_in[1];
  const int*   etype = (const int*)d_in[2];
  const int*   batch = (const int*)d_in[3];
  const float* W1    = (const float*)d_in[4];
  const float* root1 = (const float*)d_in[5];
  const float* b1    = (const float*)d_in[6];
  const float* Wl    = (const float*)d_in[7];
  const float* rootl = (const float*)d_in[8];
  const float* bl    = (const float*)d_in[9];
  const float* l1w   = (const float*)d_in[10];
  const float* l1b   = (const float*)d_in[11];
  const float* l2w   = (const float*)d_in[12];
  const float* l2b   = (const float*)d_in[13];
  float* outp = (float*)d_out;
  const int* srcv = eidx;
  const int* dstv = eidx + NEDGES;

  char* ws = (char*)d_ws;
  size_t off = 0;
  auto carve = [&](size_t bytes) -> char* {
    char* p = ws + off;
    off = (off + bytes + 255) & ~(size_t)255;
    return p;
  };
  float*          h_all  = (float*)carve((size_t)NREL * NNODES * 64 * 4);  // 102.4 MB
  float*          outbuf = (float*)carve((size_t)NNODES * 64 * 4);         // 25.6 MB
  unsigned short* afrag  = (unsigned short*)carve((size_t)NNODES * 64 * 2);// 12.8 MB
  unsigned short* wfrag  = (unsigned short*)carve((size_t)3 * 5 * 4096 * 2);
  int*            cnt    = (int*)carve((size_t)NNODES * NREL * 4);
  int*            bc     = (int*)carve((size_t)NGRAPH * 4);
  float*          pools  = (float*)carve((size_t)NGRAPH * 64 * 4);
  float*          g1     = (float*)carve((size_t)NGRAPH * 64 * 4);

  { long n = (long)NNODES * NREL; k_zero<<<(int)((n + 255) / 256), 256, 0, stream>>>((unsigned*)cnt, n); }
  { long n = NGRAPH;              k_zero<<<(int)((n + 255) / 256), 256, 0, stream>>>((unsigned*)bc, n); }
  { long n = (long)NGRAPH * 64;   k_zero<<<(int)((n + 255) / 256), 256, 0, stream>>>((unsigned*)pools, n); }

  k_count <<<(NEDGES + 255) / 256, 256, 0, stream>>>(dstv, etype, cnt, NEDGES);
  k_bcount<<<(NNODES + 255) / 256, 256, 0, stream>>>(batch, bc, NNODES);
  { int n = 3 * 5 * 4096; k_pack_w<<<(n + 255) / 256, 256, 0, stream>>>(W1, root1, Wl, rootl, wfrag); }

  const long nelem = (long)NNODES * 64;
  const int  gpack = (int)((nelem + 255) / 256);
  const int  ggemm = (5 * NBLK + 7) / 8;          // 8 waves per 256-thread block
  const long etid  = (long)NEDGES * 16;
  const int  gedge = (int)((etid + 255) / 256);

  // layer 0
  k_pack_a<<<gpack, 256, 0, stream>>>(x, afrag, outbuf, 0, 1, 0, nelem);
  k_gemm  <<<ggemm, 256, 0, stream>>>(afrag, wfrag, b1, h_all, outbuf);
  k_edges <<<gedge, 256, 0, stream>>>(srcv, dstv, etype, cnt, h_all, outbuf);
  // layer 1
  k_pack_a<<<gpack, 256, 0, stream>>>(outbuf, afrag, outbuf, 1, 1, 0, nelem);
  k_gemm  <<<ggemm, 256, 0, stream>>>(afrag, wfrag + (size_t)1 * 5 * 4096, bl + 0, h_all, outbuf);
  k_edges <<<gedge, 256, 0, stream>>>(srcv, dstv, etype, cnt, h_all, outbuf);
  // layer 2
  k_pack_a<<<gpack, 256, 0, stream>>>(outbuf, afrag, outbuf, 1, 1, 0, nelem);
  k_gemm  <<<ggemm, 256, 0, stream>>>(afrag, wfrag + (size_t)2 * 5 * 4096, bl + 64, h_all, outbuf);
  k_edges <<<gedge, 256, 0, stream>>>(srcv, dstv, etype, cnt, h_all, outbuf);
  // final ReLU in place (for pooling)
  k_pack_a<<<gpack, 256, 0, stream>>>(outbuf, afrag, outbuf, 1, 0, 1, nelem);

  { long n = (long)NNODES * 16; k_pool<<<(int)((n + 255) / 256), 256, 0, stream>>>(outbuf, batch, pools); }
  k_head1<<<(NGRAPH * 64 + 255) / 256, 256, 0, stream>>>(pools, bc, l1w, l1b, g1);
  k_head2<<<(NGRAPH * NCLS + 255) / 256, 256, 0, stream>>>(g1, l2w, l2b, outp);
}
